// SelfAttLayer_Dec_44667659878825
// MI455X (gfx1250) — matrix operand order repaired and checked by reading, not compiled, not run on hardware
//
#include <hip/hip_runtime.h>
#include <hip/hip_bf16.h>

typedef _Float16 f16;
typedef __attribute__((ext_vector_type(16))) _Float16 v16h;
typedef __attribute__((ext_vector_type(8)))  float    v8f;

typedef unsigned int u32x4 __attribute__((ext_vector_type(4)));
typedef int          i32x8 __attribute__((ext_vector_type(8)));
typedef int          i32x4 __attribute__((ext_vector_type(4)));

typedef __attribute__((address_space(1))) i32x4 as1_v4i;
typedef __attribute__((address_space(3))) i32x4 as3_v4i;
typedef __attribute__((address_space(3))) void  as3_void;

#define FFDIM 128
#define ADIM  11
#define TDIM  121
#define DDIM  256
#define HDIM  4
#define HD    64
#define BATCH (FFDIM*ADIM)     // 1408
#define TOK   (BATCH*TDIM)     // 170368
#define NQKV  (3*DDIM)         // 768

#if __has_builtin(__builtin_amdgcn_global_load_async_to_lds_b128) && __has_builtin(__builtin_amdgcn_s_wait_asynccnt)
#define HAS_ASYNC_LDS 1
#else
#define HAS_ASYNC_LDS 0
#endif

#if __has_builtin(__builtin_amdgcn_tensor_load_to_lds) && __has_builtin(__builtin_amdgcn_s_wait_tensorcnt)
#define HAS_TDM 1
#else
#define HAS_TDM 0
#endif

// ---------------- gfx1250 data movers ----------------

// 16B global -> LDS, async if available (ASYNCcnt-tracked), else through VGPRs.
__device__ __forceinline__ void copy16_to_lds(f16* lds_dst, const f16* gsrc) {
#if HAS_ASYNC_LDS
    __builtin_amdgcn_global_load_async_to_lds_b128(
        (as1_v4i*)(f16*)gsrc, (as3_v4i*)lds_dst, 0, 0);
#else
    *(uint4*)lds_dst = *(const uint4*)gsrc;
#endif
}

__device__ __forceinline__ void async_copies_wait() {
#if HAS_ASYNC_LDS
    __builtin_amdgcn_s_wait_asynccnt(0);
#endif
}

#if HAS_TDM
// 1-D TDM descriptor: nbytes (multiple of 8, fits 16-bit count of 8B elems)
// D# packing per CDNA5 ISA ch.8: group0 = {count|flags, lds_addr, ga_lo, ga_hi|type},
// group1 = {mask|data_size, dims/tiles/strides}, groups 2/3 zero (<=2D tensor).
__device__ __forceinline__ void tdm_load_1d(f16* lds_dst, const f16* gsrc, unsigned nbytes) {
    unsigned lds_off = (unsigned)(unsigned long long)(as3_void*)lds_dst;
    unsigned long long ga = (unsigned long long)(void*)(f16*)gsrc;
    unsigned ne8 = nbytes >> 3;                    // 8-byte elements
    u32x4 g0;
    g0.x = 1u;                                     // count=1 valid descriptor
    g0.y = lds_off;                                // lds_addr (bytes)
    g0.z = (unsigned)(ga & 0xffffffffu);           // global_addr[31:0]
    g0.w = (unsigned)((ga >> 32) & 0x1ffffffu) | 0x80000000u;  // ga[56:32] | type=2
    i32x8 g1;
    g1[0] = (3 << 16);                             // workgroup_mask=0, data_size=8B
    g1[1] = (int)(ne8 << 16);                      // tensor_dim0[15:0] in bits[63:48]
    g1[2] = (int)((ne8 >> 16) & 0xffffu) | (1 << 16); // tensor_dim0[31:16]; tensor_dim1=1
    g1[3] = (int)(ne8 << 16);                      // tile_dim0 in bits[127:112]
    g1[4] = 0;                                     // tile_dim1=0, tile_dim2=0
    g1[5] = (int)ne8;                              // tensor_dim0_stride[31:0]
    g1[6] = 0;
    g1[7] = 0;
    i32x4 z4; z4[0] = 0; z4[1] = 0; z4[2] = 0; z4[3] = 0;
    i32x8 z8; z8[0] = 0; z8[1] = 0; z8[2] = 0; z8[3] = 0;
    z8[4] = 0; z8[5] = 0; z8[6] = 0; z8[7] = 0;
    __builtin_amdgcn_tensor_load_to_lds(g0, g1, z4, z4, z8, 0);
}
#endif

// Stage a contiguous tile global -> LDS for the whole block, then barrier.
__device__ __forceinline__ void stage_tile(f16* lds, const f16* g, int nbytes) {
#if HAS_TDM
    if ((threadIdx.x >> 5) == 0) tdm_load_1d(lds, g, (unsigned)nbytes);  // one wave issues DMA
    __builtin_amdgcn_s_wait_tensorcnt(0);
#else
    int n16 = nbytes >> 4;
    for (int i = threadIdx.x; i < n16; i += 256) copy16_to_lds(lds + i * 8, g + i * 8);
    async_copies_wait();
#endif
    __syncthreads();
}

// ---------------- WMMA helpers (gfx1250 wave32) ----------------

__device__ __forceinline__ v8f wmma16x16x32(v16h a, v16h b, v8f c) {
    return __builtin_amdgcn_wmma_f32_16x16x32_f16(
        /*neg_a=*/false, a, /*neg_b=*/false, b,
        /*c_mod=*/(short)0, c, /*reuse_a=*/false, /*reuse_b=*/false);
}

// A fragment (16x32 f16), source row-major [row][k], stride ld halves.
__device__ __forceinline__ v16h load_a_frag(const f16* __restrict__ base, int ld, int k0) {
    int lane = threadIdx.x & 31;
    int row  = lane & 15;
    int koff = (lane >> 4) * 8;
    const f16* p = base + row * ld + k0 + koff;
    v16h a;
#pragma unroll
    for (int i = 0; i < 8; ++i) a[i] = p[i];
#pragma unroll
    for (int i = 0; i < 8; ++i) a[8 + i] = p[16 + i];
    return a;
}

// B fragment (32x16 f16) from "transposed" storage [n][k] (contiguous k per lane).
__device__ __forceinline__ v16h load_bT_frag(const f16* __restrict__ base, int ld, int k0) {
    int lane = threadIdx.x & 31;
    int col  = lane & 15;
    int kb   = k0 + (lane >> 4) * 8;
    const f16* p = base + (long)col * ld + kb;
    v16h b;
#pragma unroll
    for (int i = 0; i < 8; ++i) b[i] = p[i];
#pragma unroll
    for (int i = 0; i < 8; ++i) b[8 + i] = p[16 + i];
    return b;
}

// B fragment (32x16 f16) from row-major [k][n] storage (strided).
__device__ __forceinline__ v16h load_b_frag_rm(const f16* __restrict__ base, int ld, int k0, int n0) {
    int lane = threadIdx.x & 31;
    int col  = n0 + (lane & 15);
    int kb   = k0 + (lane >> 4) * 8;
    v16h b;
#pragma unroll
    for (int i = 0; i < 8; ++i) b[i] = base[(kb + i) * ld + col];
#pragma unroll
    for (int i = 0; i < 8; ++i) b[8 + i] = base[(kb + 16 + i) * ld + col];
    return b;
}

// ---------------- Kernel 1: fp32 -> f16 weight conversion ----------------

__global__ void k_cvt_weights(const float* __restrict__ w_in, const float* __restrict__ w_out,
                              const float* __restrict__ w_f1,
                              f16* __restrict__ wh_in, f16* __restrict__ wh_out,
                              f16* __restrict__ wh_f1) {
    int i = blockIdx.x * blockDim.x + threadIdx.x;
    if (i < NQKV * DDIM) wh_in[i] = (f16)w_in[i];
    if (i < DDIM * DDIM) {
        wh_out[i] = (f16)w_out[i];
        wh_f1[i]  = (f16)w_f1[i];
    }
}

// ---------------- Kernel 2: LayerNorm1 -> f16 (one wave per 256-wide row) --------

__global__ void k_ln1(const float* __restrict__ x, const float* __restrict__ g,
                      const float* __restrict__ bt, f16* __restrict__ xln) {
    int wave = threadIdx.x >> 5, lane = threadIdx.x & 31;
    long row = (long)blockIdx.x * 8 + wave;     // grid = TOK/8
    const float* px = x + row * DDIM;
    float v[8], s = 0.f, ss = 0.f;
#pragma unroll
    for (int i = 0; i < 8; ++i) { v[i] = px[lane + 32 * i]; s += v[i]; ss += v[i] * v[i]; }
#pragma unroll
    for (int m = 16; m >= 1; m >>= 1) { s += __shfl_xor(s, m, 32); ss += __shfl_xor(ss, m, 32); }
    float mean = s * (1.f / DDIM);
    float var  = ss * (1.f / DDIM) - mean * mean;
    float inv  = rsqrtf(var + 1e-5f);
    f16* po = xln + row * DDIM;
#pragma unroll
    for (int i = 0; i < 8; ++i) {
        int c = lane + 32 * i;
        po[c] = (f16)(((v[i] - mean) * inv) * g[c] + bt[c]);
    }
}

// ---------------- Kernel 3: QKV GEMM (M=TOK, N=768, K=256) ----------------

__global__ void k_gemm_qkv(const f16* __restrict__ xln, const f16* __restrict__ wh_in,
                           const float* __restrict__ b_in,
                           f16* __restrict__ qb, f16* __restrict__ kb, f16* __restrict__ vb) {
    __shared__ f16 As[128 * 256];                       // 64 KB
    long m0 = (long)blockIdx.x * 128;
    stage_tile(As, xln + m0 * DDIM, 128 * 256 * 2);     // TDM / async DMA

    int wave = threadIdx.x >> 5, lane = threadIdx.x & 31;
    int mrow = wave * 16;
    v16h a[8];
#pragma unroll
    for (int kc = 0; kc < 8; ++kc) a[kc] = load_a_frag(As + mrow * 256, 256, kc * 32);

    int rbase = (lane >> 4) * 8;
    int bidx[8], tidx[8];
#pragma unroll
    for (int i = 0; i < 8; ++i) {
        int m = (int)(m0 + mrow + rbase + i);
        bidx[i] = m / TDIM;
        tidx[i] = m - bidx[i] * TDIM;
    }

    for (int nt = 0; nt < NQKV / 16; ++nt) {
        int n0 = nt * 16;
        int col = lane & 15;
        float bias = b_in[n0 + col];
        v8f acc;
#pragma unroll
        for (int i = 0; i < 8; ++i) acc[i] = bias;
#pragma unroll
        for (int kc = 0; kc < 8; ++kc)
            acc = wmma16x16x32(a[kc], load_bT_frag(wh_in + (long)n0 * 256, 256, kc * 32), acc);

        int n = n0 + col;
        int which = n >> 8;                 // 0=q, 1=k, 2=v
        int r = n & 255;
        int h = r >> 6, d = r & 63;
        f16* dst = (which == 0) ? qb : (which == 1) ? kb : vb;
        float scale = (which == 0) ? 0.125f : 1.0f;   // HD^-0.5 fused into q
#pragma unroll
        for (int i = 0; i < 8; ++i) {
            long idx = (((long)(bidx[i] * HDIM + h)) * TDIM + tidx[i]) * HD + d;
            dst[idx] = (f16)(acc[i] * scale);
        }
    }
}

// ---------------- Kernel 4: attention per (batch, head) ----------------

__global__ void k_attn(const f16* __restrict__ qb, const f16* __restrict__ kb,
                       const f16* __restrict__ vb, f16* __restrict__ att) {
    __shared__ f16 Ks[128 * 64];        // 16 KB (keys zero-padded 121->128)
    __shared__ f16 Vs[128 * 64];        // 16 KB
    __shared__ f16 Ps[8 * 16 * 128];    // 32 KB (per-wave 16x128 probs)

    int bh = blockIdx.x;                 // b*4 + h
    long base = (long)bh * (TDIM * HD);

    // zero-fill pad rows 121..127 (7 rows * 64 halves = 224 dwords per matrix)
    for (int i = threadIdx.x; i < 7 * 64 / 2; i += 256) {
        ((unsigned*)(Ks + TDIM * HD))[i] = 0u;
        ((unsigned*)(Vs + TDIM * HD))[i] = 0u;
    }
    // async per-lane DMA of the valid 121 rows (121*64*2/16 = 968 x 16B)
    for (int i = threadIdx.x; i < TDIM * HD / 8; i += 256) {
        copy16_to_lds(Ks + i * 8, kb + base + i * 8);
        copy16_to_lds(Vs + i * 8, vb + base + i * 8);
    }
    async_copies_wait();
    __syncthreads();

    int wave = threadIdx.x >> 5, lane = threadIdx.x & 31;
    int qrow0 = wave * 16;

    // Q fragments (rows clamped; extra rows masked at store)
    int arow = qrow0 + (lane & 15); if (arow >= TDIM) arow = TDIM - 1;
    int koff = (lane >> 4) * 8;
    v16h aq[2];
#pragma unroll
    for (int c = 0; c < 2; ++c) {
        const f16* p = qb + base + (long)arow * HD + c * 32 + koff;
#pragma unroll
        for (int i = 0; i < 8; ++i) aq[c][i] = p[i];
#pragma unroll
        for (int i = 0; i < 8; ++i) aq[c][8 + i] = p[16 + i];
    }

    // logits: 8 key tiles of 16
    v8f lg[8];
#pragma unroll
    for (int j = 0; j < 8; ++j) {
        v8f acc;
#pragma unroll
        for (int i = 0; i < 8; ++i) acc[i] = 0.f;
#pragma unroll
        for (int c = 0; c < 2; ++c)
            acc = wmma16x16x32(aq[c], load_bT_frag(Ks + j * 16 * HD, HD, c * 32), acc);
        lg[j] = acc;
    }
    // mask padded keys (only tile 7 has keys >= 121)
    if (112 + (lane & 15) >= TDIM) {
#pragma unroll
        for (int i = 0; i < 8; ++i) lg[7][i] = -1e30f;
    }

    // softmax per row (row = i + 8*(lane>>4); keys spread over 8 tiles x 16 lanes)
    float inv_sum[8];
#pragma unroll
    for (int i = 0; i < 8; ++i) {
        float m = 0.0f;                               // add_zero_attn: logit 0 participates
#pragma unroll
        for (int j = 0; j < 8; ++j) m = fmaxf(m, lg[j][i]);
#pragma unroll
        for (int msk = 1; msk < 16; msk <<= 1) m = fmaxf(m, __shfl_xor(m, msk, 32));
        float s = 0.f;
#pragma unroll
        for (int j = 0; j < 8; ++j) { float e = __expf(lg[j][i] - m); lg[j][i] = e; s += e; }
#pragma unroll
        for (int msk = 1; msk < 16; msk <<= 1) s += __shfl_xor(s, msk, 32);
        s += __expf(0.f - m);                         // the single zero-attn term
        inv_sum[i] = 1.0f / s;
    }

    // P -> LDS (f16), then reload in A-fragment layout
    f16* pw = Ps + wave * 16 * 128;
    {
        int r0 = (lane >> 4) * 8;
        int c0 = lane & 15;
#pragma unroll
        for (int j = 0; j < 8; ++j)
#pragma unroll
            for (int i = 0; i < 8; ++i)
                pw[(r0 + i) * 128 + j * 16 + c0] = (f16)(lg[j][i] * inv_sum[i]);
    }
    __syncthreads();

    v16h ap[4];
#pragma unroll
    for (int c = 0; c < 4; ++c) ap[c] = load_a_frag(pw, 128, c * 32);

    int b = bh >> 2, h = bh & 3;
#pragma unroll
    for (int n = 0; n < 4; ++n) {
        v8f acc;
#pragma unroll
        for (int i = 0; i < 8; ++i) acc[i] = 0.f;
#pragma unroll
        for (int c = 0; c < 4; ++c)
            acc = wmma16x16x32(ap[c], load_b_frag_rm(Vs, HD, c * 32, n * 16), acc);
        int r0 = (lane >> 4) * 8;
        int col = n * 16 + (lane & 15);
#pragma unroll
        for (int i = 0; i < 8; ++i) {
            int t = qrow0 + r0 + i;
            if (t < TDIM)
                att[((long)(b * TDIM + t)) * DDIM + h * HD + col] = (f16)acc[i];
        }
    }
}

// ---------------- Kernel 5: out-proj + bias + residual (s = att@W_o^T + b + x) ----

__global__ void k_gemm_out(const f16* __restrict__ att, const f16* __restrict__ wh_out,
                           const float* __restrict__ b_out, const float* __restrict__ x,
                           f16* __restrict__ s) {
    __shared__ f16 As[128 * 256];
    long m0 = (long)blockIdx.x * 128;
    stage_tile(As, att + m0 * DDIM, 128 * 256 * 2);

    int wave = threadIdx.x >> 5, lane = threadIdx.x & 31;
    int mrow = wave * 16;
    v16h a[8];
#pragma unroll
    for (int kc = 0; kc < 8; ++kc) a[kc] = load_a_frag(As + mrow * 256, 256, kc * 32);

    int rbase = (lane >> 4) * 8;
    for (int nt = 0; nt < 16; ++nt) {
        int n0 = nt * 16;
        int col = lane & 15;
        float bias = b_out[n0 + col];
        v8f acc;
#pragma unroll
        for (int i = 0; i < 8; ++i) acc[i] = bias;
#pragma unroll
        for (int kc = 0; kc < 8; ++kc)
            acc = wmma16x16x32(a[kc], load_bT_frag(wh_out + (long)n0 * 256, 256, kc * 32), acc);
#pragma unroll
        for (int i = 0; i < 8; ++i) {
            long m = m0 + mrow + rbase + i;
            float val = acc[i] + x[m * DDIM + n0 + col];
            s[m * DDIM + n0 + col] = (f16)val;
        }
    }
}

// ---------------- Kernel 6: FFN GEMM + ReLU + LayerNorm2 -> fp32 output ----------

__global__ void k_f1_ln(const f16* __restrict__ s, const f16* __restrict__ wh_f1,
                        const float* __restrict__ b_f1, const float* __restrict__ g2,
                        const float* __restrict__ b2, float* __restrict__ out) {
    __shared__ f16  As[32 * 256];     // 16 KB
    __shared__ float Fs[32 * 256];    // 32 KB
    long m0 = (long)blockIdx.x * 32;
    stage_tile(As, s + m0 * DDIM, 32 * 256 * 2);

    int wave = threadIdx.x >> 5, lane = threadIdx.x & 31;
    int mt  = wave & 1;               // which 16-row half
    int ntg = (wave >> 1) * 4;        // 4 N-tiles per wave
    v16h a[8];
#pragma unroll
    for (int kc = 0; kc < 8; ++kc) a[kc] = load_a_frag(As + mt * 16 * 256, 256, kc * 32);

    int rbase = (lane >> 4) * 8;
    for (int nn = 0; nn < 4; ++nn) {
        int n0 = (ntg + nn) * 16;
        int col = lane & 15;
        float bias = b_f1[n0 + col];
        v8f acc;
#pragma unroll
        for (int i = 0; i < 8; ++i) acc[i] = bias;
#pragma unroll
        for (int kc = 0; kc < 8; ++kc)
            acc = wmma16x16x32(a[kc], load_bT_frag(wh_f1 + (long)n0 * 256, 256, kc * 32), acc);
#pragma unroll
        for (int i = 0; i < 8; ++i)
            Fs[(mt * 16 + rbase + i) * 256 + n0 + col] = fmaxf(acc[i], 0.f);
    }
    __syncthreads();

    // LayerNorm over 256: 8 threads per row, 32 elems each
    int row = threadIdx.x >> 3, sub = threadIdx.x & 7;
    const float* pr = Fs + row * 256;
    float sum = 0.f, ss = 0.f;
#pragma unroll
    for (int i = 0; i < 32; ++i) { float v = pr[sub * 32 + i]; sum += v; ss += v * v; }
#pragma unroll
    for (int msk = 1; msk < 8; msk <<= 1) { sum += __shfl_xor(sum, msk, 32); ss += __shfl_xor(ss, msk, 32); }
    float mean = sum * (1.f / DDIM);
    float var  = ss * (1.f / DDIM) - mean * mean;
    float inv  = rsqrtf(var + 1e-5f);
    float* po = out + (m0 + row) * DDIM;
#pragma unroll
    for (int i = 0; i < 32; ++i) {
        int c = sub * 32 + i;
        po[c] = (pr[c] - mean) * inv * g2[c] + b2[c];
    }
}

// ---------------- Host launcher ----------------

extern "C" void kernel_launch(void* const* d_in, const int* in_sizes, int n_in,
                              void* d_out, int out_size, void* d_ws, size_t ws_size,
                              hipStream_t stream) {
    (void)in_sizes; (void)n_in; (void)out_size; (void)ws_size;

    const float* x     = (const float*)d_in[0];
    // d_in[1] = batch_mask (unused: across_time attention)
    const float* ln1_g = (const float*)d_in[2];
    const float* ln1_b = (const float*)d_in[3];
    const float* w_in  = (const float*)d_in[4];
    const float* b_in  = (const float*)d_in[5];
    const float* w_out = (const float*)d_in[6];
    const float* b_out = (const float*)d_in[7];
    const float* w_f1  = (const float*)d_in[8];
    const float* b_f1  = (const float*)d_in[9];
    const float* ln2_g = (const float*)d_in[10];
    const float* ln2_b = (const float*)d_in[11];
    float* out = (float*)d_out;

    char* ws = (char*)d_ws;
    const size_t act_bytes = (size_t)TOK * DDIM * sizeof(f16);   // 87.2 MB
    f16* wh_in  = (f16*)(ws);
    f16* wh_out = (f16*)(ws + 393216);
    f16* wh_f1  = (f16*)(ws + 524288);
    f16* xln    = (f16*)(ws + 655360);               // aliased later by att
    f16* qbuf   = (f16*)(ws + 655360 + act_bytes);   // aliased later by s
    f16* kbuf   = (f16*)(ws + 655360 + 2 * act_bytes);
    f16* vbuf   = (f16*)(ws + 655360 + 3 * act_bytes);
    f16* attb   = xln;    // xln dead after QKV GEMM
    f16* sbuf   = qbuf;   // q dead after attention

    k_cvt_weights<<<768, 256, 0, stream>>>(w_in, w_out, w_f1, wh_in, wh_out, wh_f1);
    k_ln1<<<TOK / 8, 256, 0, stream>>>(x, ln1_g, ln1_b, xln);
    k_gemm_qkv<<<TOK / 128, 256, 0, stream>>>(xln, wh_in, b_in, qbuf, kbuf, vbuf);
    k_attn<<<BATCH * HDIM, 256, 0, stream>>>(qbuf, kbuf, vbuf, attb);
    k_gemm_out<<<TOK / 128, 256, 0, stream>>>(attb, wh_out, b_out, x, sbuf);
    k_f1_ln<<<TOK / 32, 256, 0, stream>>>(sbuf, wh_f1, b_f1, ln2_g, ln2_b, out);
}
